// AssociativeWithRpeOperator_5531917878023
// MI455X (gfx1250) — matrix operand order repaired and checked by reading, not compiled, not run on hardware
//
#include <hip/hip_runtime.h>

typedef __attribute__((ext_vector_type(16))) _Float16 v16h;
typedef __attribute__((ext_vector_type(8)))  float    v8f;

#define H_    12
#define HS_   64
#define ALL_  768
#define P_    128
#define B_    2
#define S_    1024
#define NPAD  272                  // 2P+2 = 258 padded to 17*16
#define INV_SCAL 0.35355339059327373f   // 1 / 64^0.25

// ---- WMMA fragment helpers (wave32, 16x16x32 f16 -> f32) -------------------
// A (16xK=32 f16), per ISA 7.12.2: lanes 0-15 row M=lane: halfs 0..7 = K0..7,
// halfs 8..15 = K16..23; lanes 16-31 row M=lane-16: K8..15 / K24..31.
__device__ __forceinline__ v16h load_a16(const _Float16* A, int lda, int row0,
                                         int k0, int lane) {
  const _Float16* p = A + (size_t)(row0 + (lane & 15)) * lda + k0 + ((lane >> 4) << 3);
  v16h a;
#pragma unroll
  for (int i = 0; i < 8; ++i) { a[i] = p[i]; a[i + 8] = p[i + 16]; }
  return a;
}
// B (K=32 x 16 f16) loaded from an N-major array Bt[N][K]: lane holds one
// column n = lane&15; lanes 0-15 hold K0..7 / K16..23, lanes 16-31 hold K8..15 / K24..31.
__device__ __forceinline__ v16h load_bt16(const _Float16* Bt, int ldb, int n0,
                                          int k0, int lane) {
  const _Float16* p = Bt + (size_t)(n0 + (lane & 15)) * ldb + k0 + ((lane >> 4) << 3);
  v16h b;
#pragma unroll
  for (int i = 0; i < 8; ++i) { b[i] = p[i]; b[i + 8] = p[i + 16]; }
  return b;
}
__device__ __forceinline__ v8f wmma_f16(v16h a, v16h b, v8f c) {
  return __builtin_amdgcn_wmma_f32_16x16x32_f16(false, a, false, b, (short)0, c,
                                                false, false);
}

// ---- kernel 1: f32 -> f16 conversions / transposes ------------------------
__global__ void k_prep(const float* __restrict__ p1, const float* __restrict__ Wqk,
                       const float* __restrict__ rel, _Float16* __restrict__ p1h,
                       _Float16* __restrict__ wqkt, _Float16* __restrict__ relh) {
  int i = blockIdx.x * blockDim.x + threadIdx.x;
  const int NP1 = B_ * S_ * ALL_;          // 1,572,864
  if (i < NP1) p1h[i] = (_Float16)p1[i];
  const int NW = ALL_ * 2 * ALL_;          // 1,179,648
  if (i < NW) {                            // store W_qk transposed: [N=1536][K=768]
    int n = i % (2 * ALL_), kd = i / (2 * ALL_);
    wqkt[(size_t)n * ALL_ + kd] = (_Float16)Wqk[i];
  }
  const int NR = NPAD * HS_;               // 17,408
  if (i < NR) {                            // rel_emb pre-scaled by 1/scaling
    int j = i / HS_;
    relh[i] = (j < 2 * P_ + 2) ? (_Float16)(rel[i] * INV_SCAL) : (_Float16)0.f;
  }
}

// ---- kernel 2: u = p1 @ W_qk + b_qk ; split/scale into q,k (f16) -----------
__global__ void k_qkproj(const _Float16* __restrict__ p1h,
                         const _Float16* __restrict__ wqkt,
                         const float* __restrict__ bqk,
                         _Float16* __restrict__ qh, _Float16* __restrict__ kh) {
  const int lane = threadIdx.x & 31;
  const int tid = blockIdx.x * (blockDim.x >> 5) + (threadIdx.x >> 5);
  const int m_tile = tid / 96;             // M = 2048 -> 128 tiles
  const int n_tile = tid % 96;             // N = 1536 -> 96 tiles
  v8f acc = {0.f, 0.f, 0.f, 0.f, 0.f, 0.f, 0.f, 0.f};
#pragma unroll 4
  for (int k = 0; k < ALL_; k += 32) {     // K = 768 -> 24 WMMA steps
    v16h a = load_a16(p1h, ALL_, m_tile * 16, k, lane);
    v16h b = load_bt16(wqkt, ALL_, n_tile * 16, k, lane);
    acc = wmma_f16(a, b, acc);
  }
  const int gn = n_tile * 16 + (lane & 15);
  const float bias = bqk[gn];
#pragma unroll
  for (int i = 0; i < 8; ++i) {
    int gm = m_tile * 16 + i + ((lane >> 4) << 3);
    float v = (acc[i] + bias) * INV_SCAL;
    int bb = gm >> 10, s = gm & 1023;
    if (gn < ALL_) {
      int h = gn >> 6, d = gn & 63;
      qh[(size_t)(((bb * H_ + h) << 10) + s) * HS_ + d] = (_Float16)v;
    } else {
      int n2 = gn - ALL_;
      int h = n2 >> 6, d = n2 & 63;
      kh[(size_t)(((bb * H_ + h) << 10) + s) * HS_ + d] = (_Float16)v;
    }
  }
}

// ---- kernel 3: r[b,h,s,j] = q . rel_emb[j]  (bias gather table) ------------
__global__ void k_rbias(const _Float16* __restrict__ qh,
                        const _Float16* __restrict__ relh,
                        float* __restrict__ rbuf) {
  const int lane = threadIdx.x & 31;
  const int tid = blockIdx.x * (blockDim.x >> 5) + (threadIdx.x >> 5);
  const int m_tile = tid / 17;             // M = 24576 -> 1536 tiles
  const int n_tile = tid % 17;             // N = 272   -> 17 tiles
  v8f acc = {0.f, 0.f, 0.f, 0.f, 0.f, 0.f, 0.f, 0.f};
  acc = wmma_f16(load_a16(qh, HS_, m_tile * 16, 0, lane),
                 load_bt16(relh, HS_, n_tile * 16, 0, lane), acc);
  acc = wmma_f16(load_a16(qh, HS_, m_tile * 16, 32, lane),
                 load_bt16(relh, HS_, n_tile * 16, 32, lane), acc);
  const int gn = n_tile * 16 + (lane & 15);
#pragma unroll
  for (int i = 0; i < 8; ++i) {
    int gm = m_tile * 16 + i + ((lane >> 4) << 3);
    rbuf[(size_t)gm * NPAD + gn] = acc[i];
  }
}

// ---- kernel 4: fused scores + toeplitz bias + head-mix + output ------------
// block = 128 threads (4 waves); each wave owns one 16x16 (s,t) tile.
__global__ __launch_bounds__(128) void k_attn_out(
    const _Float16* __restrict__ qh, const _Float16* __restrict__ kh,
    const float* __restrict__ rbuf, const float* __restrict__ Wout,
    const float* __restrict__ bout, float* __restrict__ out) {
  __shared__ float lds_sc[4 * 256 * 12];   // per-wave [pos][head]
  __shared__ float lds_w[12 * 64];
  __shared__ float lds_b[64];
  const int lane = threadIdx.x & 31;
  const int wid = threadIdx.x >> 5;

  for (int i = threadIdx.x; i < 12 * 64; i += blockDim.x) lds_w[i] = Wout[i];
  if (threadIdx.x < 64) lds_b[threadIdx.x] = bout[threadIdx.x];
  __syncthreads();

  const int bb = blockIdx.z;
  const int s0 = blockIdx.y * 16;
  const int t0 = blockIdx.x * 64 + wid * 16;
  const int tq = lane & 15;
  const int mrow_base = (lane >> 4) << 3;

  v8f acc[12];
#pragma unroll
  for (int h = 0; h < H_; ++h) {
    const _Float16* qb = qh + (size_t)((bb * H_ + h) << 10) * HS_;
    const _Float16* kb = kh + (size_t)((bb * H_ + h) << 10) * HS_;
    v8f c = {0.f, 0.f, 0.f, 0.f, 0.f, 0.f, 0.f, 0.f};
    c = wmma_f16(load_a16(qb, HS_, s0, 0, lane), load_bt16(kb, HS_, t0, 0, lane), c);
    c = wmma_f16(load_a16(qb, HS_, s0, 32, lane), load_bt16(kb, HS_, t0, 32, lane), c);
    // relative-position bias: coalesced Toeplitz gather (consecutive t ->
    // consecutive ids -> consecutive rbuf entries within a lane row)
    const float* rrow = rbuf + (size_t)((bb * H_ + h) << 10) * NPAD;
#pragma unroll
    for (int i = 0; i < 8; ++i) {
      int s = s0 + i + mrow_base;
      int t = t0 + tq;
      int d = t - s;
      int idx = (d >= 0) ? (d < 127 ? d : 127)
                         : (256 - ((-d) < 127 ? (-d) : 127));
      c[i] += rrow[(size_t)s * NPAD + idx];
    }
    acc[h] = c;
  }

  // transpose WMMA C-layout -> [pos][head] through LDS
  float* myLds = &lds_sc[wid * 256 * 12];
#pragma unroll
  for (int h = 0; h < H_; ++h) {
#pragma unroll
    for (int i = 0; i < 8; ++i) {
      int m = i + mrow_base;
      myLds[(m * 16 + tq) * 12 + h] = acc[h][i];
    }
  }
  __syncthreads();

  // head-mix 12 -> 64 and fully-coalesced float4 output stores
  for (int p = lane; p < 256; p += 32) {
    float sc[12];
#pragma unroll
    for (int h = 0; h < 12; ++h) sc[h] = myLds[p * 12 + h];
    int m = p >> 4, n = p & 15;
    int s = s0 + m, t = t0 + n;
    float* op = out + (((size_t)(bb * S_ + s)) * S_ + t) * 64;
#pragma unroll
    for (int c = 0; c < 64; c += 4) {
      float v0 = lds_b[c], v1 = lds_b[c + 1], v2 = lds_b[c + 2], v3 = lds_b[c + 3];
#pragma unroll
      for (int h = 0; h < 12; ++h) {
        float sh = sc[h];
        v0 += sh * lds_w[h * 64 + c];
        v1 += sh * lds_w[h * 64 + c + 1];
        v2 += sh * lds_w[h * 64 + c + 2];
        v3 += sh * lds_w[h * 64 + c + 3];
      }
      float4 v; v.x = v0; v.y = v1; v.z = v2; v.w = v3;
      *(float4*)(op + c) = v;
    }
  }
}

extern "C" void kernel_launch(void* const* d_in, const int* in_sizes, int n_in,
                              void* d_out, int out_size, void* d_ws, size_t ws_size,
                              hipStream_t stream) {
  (void)in_sizes; (void)n_in; (void)out_size; (void)ws_size;
  const float* p1   = (const float*)d_in[1];
  const float* Wqk  = (const float*)d_in[3];
  const float* bqk  = (const float*)d_in[4];
  const float* rel  = (const float*)d_in[5];
  const float* Wout = (const float*)d_in[6];
  const float* bout = (const float*)d_in[7];
  float* out = (float*)d_out;

  char* ws = (char*)d_ws;
  size_t off = 0;
  _Float16* p1h  = (_Float16*)(ws + off); off += (size_t)B_ * S_ * ALL_ * 2;       // 3.00 MB
  _Float16* wqkt = (_Float16*)(ws + off); off += (size_t)ALL_ * 2 * ALL_ * 2;      // 2.25 MB
  _Float16* relh = (_Float16*)(ws + off); off += (size_t)NPAD * HS_ * 2;           // 34 KB
  _Float16* qh   = (_Float16*)(ws + off); off += (size_t)B_ * H_ * S_ * HS_ * 2;   // 3.00 MB
  _Float16* kh   = (_Float16*)(ws + off); off += (size_t)B_ * H_ * S_ * HS_ * 2;   // 3.00 MB
  float*    rbuf = (float*)(ws + off);    off += (size_t)B_ * H_ * S_ * NPAD * 4;  // 25.5 MB

  const int nprep = ALL_ * 2 * ALL_;  // W_qk range only; the p1 range NP1 is larger
  hipLaunchKernelGGL(k_prep, dim3((nprep + 255) / 256), dim3(256), 0, stream,
                     p1, Wqk, rel, p1h, wqkt, relh);
  hipLaunchKernelGGL(k_qkproj, dim3(12288 / 8), dim3(256), 0, stream,
                     p1h, wqkt, bqk, qh, kh);
  hipLaunchKernelGGL(k_rbias, dim3(26112 / 8), dim3(256), 0, stream,
                     qh, relh, rbuf);
  hipLaunchKernelGGL(k_attn_out, dim3(16, 64, 2), dim3(128), 0, stream,
                     qh, kh, rbuf, Wout, bout, out);
}
